// GraphTL_89404039234075
// MI455X (gfx1250) — matrix operand order repaired and checked by reading, not compiled, not run on hardware
//
#include <hip/hip_runtime.h>

// ---------------------------------------------------------------------------
// Problem constants (match reference)
// ---------------------------------------------------------------------------
#define N_NODES 6144
#define NH1 4
#define NH2 6
#define SLOPE 0.2f
#define BN_EPS 1e-5f
#define INV_TEMP 2.0f   // 1 / TEMP, TEMP = 0.5

// ---------------------------------------------------------------------------
// bf16 helpers (raw u16 storage; avoids __hip_bfloat16 ctor issues in LDS)
// ---------------------------------------------------------------------------
typedef __bf16 v16bf __attribute__((ext_vector_type(16)));
typedef float  v8f   __attribute__((ext_vector_type(8)));
typedef unsigned int u32x4 __attribute__((ext_vector_type(4)));
typedef int i32x8 __attribute__((ext_vector_type(8)));
typedef int i32x4 __attribute__((ext_vector_type(4)));

struct alignas(16) I4 { int x, y, z, w; };
union FragU { I4 q[2]; v16bf v; };

__device__ __forceinline__ unsigned short f2bf(float f) {
  unsigned int u = __float_as_uint(f);
  u += 0x7FFFu + ((u >> 16) & 1u);      // round-to-nearest-even
  return (unsigned short)(u >> 16);
}
__device__ __forceinline__ float bf2f(unsigned short h) {
  return __uint_as_float(((unsigned int)h) << 16);
}

// A-fragment (16x32 bf16, ISA 7.12.2): lane L holds row m=L%16,
// K chunks [c0,c0+8) and [c0+16,c0+24) with c0 = 8*(L/16).
__device__ __forceinline__ v16bf frag_a(const unsigned short* base, int stride, int lane) {
  int r = lane & 15, hf = lane >> 4;
  const unsigned short* p = base + (size_t)r * stride + hf * 8;
  FragU u;
  u.q[0] = *(const I4*)(p);
  u.q[1] = *(const I4*)(p + 16);
  return u.v;
}
// B-fragment (32x16 bf16): lane L holds col n=L%16, K chunks [c0,c0+8) and
// [c0+16,c0+24) with c0 = 8*(L/16). base must point at (col-major) row n stride `stride`.
__device__ __forceinline__ v16bf frag_b(const unsigned short* base, int stride, int lane) {
  int r = lane & 15, hf = lane >> 4;
  const unsigned short* p = base + (size_t)r * stride + hf * 8;
  FragU u;
  u.q[0] = *(const I4*)(p);
  u.q[1] = *(const I4*)(p + 16);
  return u.v;
}
__device__ __forceinline__ v8f wmma_bf16(v16bf a, v16bf b, v8f c) {
  return __builtin_amdgcn_wmma_f32_16x16x32_bf16(false, a, false, b, (short)0, c,
                                                 false, false);
}

// ---------------------------------------------------------------------------
// Tensor Data Mover: DMA a 2-D tile (tile_k contiguous elems x tile_m rows of
// bf16, row stride = stride_elems) from global into LDS at lds_off, inserting
// 4 DWORDs of padding after every 16 DWORDs (64B row -> 80B LDS row stride,
// i.e. the [*][40] u16 layout used by the fragment loaders).
// D# layout per CDNA5 ISA section 8 (group0 128b + group1 256b, groups 2/3
// zero => 2-D tensor).
// ---------------------------------------------------------------------------
__device__ __forceinline__ void tdm_load_tile_2d(const unsigned short* gsrc,
                                                 unsigned lds_off,
                                                 unsigned tile_k, unsigned tile_m,
                                                 unsigned stride_elems) {
  unsigned long long ga = (unsigned long long)gsrc;
  u32x4 g0;
  g0[0] = 1u;                                                // count=1 (user D#)
  g0[1] = lds_off;                                           // lds_addr (bytes)
  g0[2] = (unsigned)(ga & 0xFFFFFFFFu);                      // global_addr[31:0]
  g0[3] = (unsigned)((ga >> 32) & 0x01FFFFFFu) | (2u << 30); // addr[56:32], type=2
  i32x8 g1;
  g1[0] = (int)((1u << 16)      // data_size = 2 bytes
              | (1u << 20)      // pad_enable
              | (3u << 22)      // pad_interval: 16 DWORDs
              | (3u << 25));    // pad_amount:   4 DWORDs
  g1[1] = (int)((tile_k & 0xFFFFu) << 16);                         // tensor_dim0 lo
  g1[2] = (int)(((tile_k >> 16) & 0xFFFFu) | ((tile_m & 0xFFFFu) << 16)); // dim0 hi | dim1 lo
  g1[3] = (int)(((tile_m >> 16) & 0xFFFFu) | ((tile_k & 0xFFFFu) << 16)); // dim1 hi | tile_dim0
  g1[4] = (int)(tile_m & 0xFFFFu);                                 // tile_dim1 (tile_dim2=0)
  g1[5] = (int)stride_elems;                                       // tensor_dim0_stride lo
  g1[6] = 0;                                                       // stride hi | dim1_stride lo
  g1[7] = 0;
  i32x4 z4 = {0, 0, 0, 0};
#if __clang_major__ >= 23
  i32x8 z8 = {0, 0, 0, 0, 0, 0, 0, 0};
  __builtin_amdgcn_tensor_load_to_lds(g0, g1, z4, z4, z8, 0);
#else
  __builtin_amdgcn_tensor_load_to_lds(g0, g1, z4, z4, 0);
#endif
}

__device__ __forceinline__ void atomicMaxF(float* addr, float val) {
  if (val >= 0.f) atomicMax((int*)addr, __float_as_int(val));
  else            atomicMin((unsigned int*)addr, __float_as_uint(val));
}

__device__ __forceinline__ float eluf(float x) {
  return x > 0.f ? x : __expf(x) - 1.f;
}

// ---------------------------------------------------------------------------
// Elementwise / init kernels
// ---------------------------------------------------------------------------
__global__ void k_cvt_bf16(const float* __restrict__ in,
                           unsigned short* __restrict__ out, int n) {
  int t = blockIdx.x * blockDim.x + threadIdx.x;
  if (t < n) out[t] = f2bf(in[t]);
}

__global__ void k_fill(float* __restrict__ p, float v, int n) {
  int t = blockIdx.x * blockDim.x + threadIdx.x;
  if (t < n) p[t] = v;
}

// ---------------------------------------------------------------------------
// bf16 WMMA GEMM: C[M,Nc] = A[M,K] @ B[K,Nc] (+ bias[Nc])
// block tile 128x128, K-step 32, 8 waves of (4x2) 16x16 accumulators.
// A tile staged by the Tensor Data Mover (wave 0 issues tensor_load_to_lds,
// TENSORcnt-synchronized); B tile manually transposed into LDS.
// M, Nc multiples of 128; K multiple of 32.
// ---------------------------------------------------------------------------
__global__ __launch_bounds__(256) void k_gemm(
    const unsigned short* __restrict__ A,
    const unsigned short* __restrict__ B,
    float* __restrict__ C,
    const float* __restrict__ bias,
    int K, int Nc)
{
  __shared__ alignas(16) unsigned short As[128][40];  // [m][k] (TDM-padded rows)
  __shared__ alignas(16) unsigned short Bs[128][40];  // [n][k] (transposed)
  int tid  = threadIdx.x;
  int lane = tid & 31, w = tid >> 5;
  int bm = blockIdx.y * 128, bn = blockIdx.x * 128;
  int wm = (w >> 2) * 64;          // wave covers 64 rows
  int wn = (w & 3) * 32;           // wave covers 32 cols

  // LDS byte offset of As: generic LDS addresses carry the byte offset in
  // their low 32 bits (ISA 10.2 aperture mapping).
  unsigned ldsA = (unsigned)(unsigned long long)(void*)&As[0][0];

  const v8f vz = {0.f, 0.f, 0.f, 0.f, 0.f, 0.f, 0.f, 0.f};
  v8f acc[4][2];
#pragma unroll
  for (int i = 0; i < 4; ++i)
#pragma unroll
    for (int j = 0; j < 2; ++j) acc[i][j] = vz;

  for (int k0 = 0; k0 < K; k0 += 32) {
    __syncthreads();   // previous iteration's fragment reads complete
    if (w == 0)        // async DMA of the 128x32 A tile into padded LDS rows
      tdm_load_tile_2d(A + (size_t)bm * K + k0, ldsA, 32u, 128u, (unsigned)K);
    {   // B tile, stored transposed: thread loads 16 contiguous n's for one k
      int kk = tid >> 3, noff = (tid & 7) * 16;
      const unsigned short* s = B + (size_t)(k0 + kk) * Nc + bn + noff;
#pragma unroll
      for (int q = 0; q < 16; ++q) Bs[noff + q][kk] = s[q];
      if (k0 + 32 < K) __builtin_prefetch(s + (size_t)32 * Nc, 0, 1);
    }
    if (w == 0) __builtin_amdgcn_s_wait_tensorcnt(0);
    __syncthreads();   // A tile (TDM) + B tile (ds stores) visible to all waves

    v16bf fa[4], fb[2];
#pragma unroll
    for (int i = 0; i < 4; ++i) fa[i] = frag_a(&As[wm + i * 16][0], 40, lane);
#pragma unroll
    for (int j = 0; j < 2; ++j) fb[j] = frag_b(&Bs[wn + j * 16][0], 40, lane);
#pragma unroll
    for (int i = 0; i < 4; ++i)
#pragma unroll
      for (int j = 0; j < 2; ++j) acc[i][j] = wmma_bf16(fa[i], fb[j], acc[i][j]);
  }

  // C/D layout: lane holds col n=L%16, rows m = r + 8*(L/16)
  int hf = lane >> 4, n = lane & 15;
#pragma unroll
  for (int i = 0; i < 4; ++i)
#pragma unroll
    for (int j = 0; j < 2; ++j) {
      int gm0 = bm + wm + i * 16 + hf * 8;
      int gn  = bn + wn + j * 16 + n;
      float bv = bias ? bias[gn] : 0.f;
#pragma unroll
      for (int r = 0; r < 8; ++r)
        C[(size_t)(gm0 + r) * Nc + gn] = acc[i][j][r] + bv;
    }
}

// ---------------------------------------------------------------------------
// GAT attention pieces
// ---------------------------------------------------------------------------
// el[n,h] = sum_d F[n,h,d]*al[h,d] ; er likewise. blockDim = 32*H, grid = N.
__global__ void k_elr(const float* __restrict__ F,
                      const float* __restrict__ al,
                      const float* __restrict__ ar,
                      float* __restrict__ el, float* __restrict__ er, int H) {
  int n = blockIdx.x;
  int h = threadIdx.x >> 5, lane = threadIdx.x & 31;
  const float* f = F + (size_t)n * H * 256 + h * 256;
  float sl = 0.f, sr = 0.f;
  for (int d = lane; d < 256; d += 32) {
    float x = f[d];
    sl += x * al[h * 256 + d];
    sr += x * ar[h * 256 + d];
  }
#pragma unroll
  for (int m = 16; m >= 1; m >>= 1) {
    sl += __shfl_xor(sl, m, 32);
    sr += __shfl_xor(sr, m, 32);
  }
  if (lane == 0) { el[n * H + h] = sl; er[n * H + h] = sr; }
}

__global__ void k_edge_max(const float* __restrict__ el, const float* __restrict__ er,
                           const int* __restrict__ src, const int* __restrict__ dst,
                           float* __restrict__ emax, int Etot, int H) {
  int t = blockIdx.x * blockDim.x + threadIdx.x;
  if (t >= Etot * H) return;
  int e = t / H, h = t - e * H;
  float v = el[src[e] * H + h] + er[dst[e] * H + h];
  v = v >= 0.f ? v : SLOPE * v;
  atomicMaxF(&emax[dst[e] * H + h], v);
}

__global__ void k_edge_den(const float* __restrict__ el, const float* __restrict__ er,
                           const int* __restrict__ src, const int* __restrict__ dst,
                           const float* __restrict__ emax, float* __restrict__ den,
                           int Etot, int H) {
  int t = blockIdx.x * blockDim.x + threadIdx.x;
  if (t >= Etot * H) return;
  int e = t / H, h = t - e * H;
  int di = dst[e];
  float v = el[src[e] * H + h] + er[di * H + h];
  v = v >= 0.f ? v : SLOPE * v;
  atomicAdd(&den[di * H + h], __expf(v - emax[di * H + h]));
}

// grid = Etot, block = 256 (one dim per thread); O += a * F[src]
__global__ void k_edge_agg(const float* __restrict__ el, const float* __restrict__ er,
                           const float* __restrict__ emax, const float* __restrict__ den,
                           const int* __restrict__ src, const int* __restrict__ dst,
                           const float* __restrict__ F, float* __restrict__ O, int H) {
  int e = blockIdx.x;
  int si = src[e], di = dst[e];
  int d = threadIdx.x;
  int HD = H * 256;
  for (int h = 0; h < H; ++h) {
    float v = el[si * H + h] + er[di * H + h];
    v = v >= 0.f ? v : SLOPE * v;
    float a = __expf(v - emax[di * H + h]) / den[di * H + h];
    atomicAdd(&O[(size_t)di * HD + h * 256 + d],
              a * F[(size_t)si * HD + h * 256 + d]);
  }
}

// ---------------------------------------------------------------------------
// BatchNorm (1024 features) + ELU -> bf16
// ---------------------------------------------------------------------------
__global__ void k_bn_stats(const float* __restrict__ O,
                           float* __restrict__ sum, float* __restrict__ ss) {
  int n0 = blockIdx.x * 4;
#pragma unroll
  for (int q = 0; q < 4; ++q) {
    int f = threadIdx.x + q * 256;
    float s = 0.f, s2 = 0.f;
#pragma unroll
    for (int r = 0; r < 4; ++r) {
      float x = O[(size_t)(n0 + r) * 1024 + f];
      s += x; s2 += x * x;
    }
    atomicAdd(&sum[f], s);
    atomicAdd(&ss[f], s2);
  }
}

__global__ void k_bn_apply(const float* __restrict__ O,
                           const float* __restrict__ sum, const float* __restrict__ ss,
                           const float* __restrict__ g, const float* __restrict__ be,
                           unsigned short* __restrict__ out) {
  int n = blockIdx.x;
  const float invN = 1.f / (float)N_NODES;
#pragma unroll
  for (int q = 0; q < 4; ++q) {
    int f = threadIdx.x + q * 256;
    float m = sum[f] * invN;
    float var = ss[f] * invN - m * m;
    float x = O[(size_t)n * 1024 + f];
    float y = (x - m) * rsqrtf(var + BN_EPS) * g[f] + be[f];
    out[(size_t)n * 1024 + f] = f2bf(eluf(y));
  }
}

// ELU then mean over H2 heads -> bf16 [N,256]
__global__ void k_elu_mean(const float* __restrict__ O2,
                           unsigned short* __restrict__ out) {
  int n = blockIdx.x, d = threadIdx.x;
  float s = 0.f;
#pragma unroll
  for (int h = 0; h < NH2; ++h) s += eluf(O2[(size_t)n * (NH2 * 256) + h * 256 + d]);
  out[(size_t)n * 256 + d] = f2bf(s * (1.f / (float)NH2));
}

// row L2-normalize [N,256] f32 -> bf16
__global__ __launch_bounds__(256) void k_l2norm(const float* __restrict__ Z,
                                                unsigned short* __restrict__ out) {
  __shared__ float red[8];
  int n = blockIdx.x, d = threadIdx.x;
  float v = Z[(size_t)n * 256 + d];
  float s = v * v;
#pragma unroll
  for (int m = 16; m >= 1; m >>= 1) s += __shfl_xor(s, m, 32);
  if ((d & 31) == 0) red[d >> 5] = s;
  __syncthreads();
  if (d == 0) {
    float t = 0.f;
#pragma unroll
    for (int i = 0; i < 8; ++i) t += red[i];
    red[0] = 1.f / fmaxf(sqrtf(t), 1e-12f);
  }
  __syncthreads();
  out[(size_t)n * 256 + d] = f2bf(v * red[0]);
}

// ---------------------------------------------------------------------------
// Fused contrastive loss: per wave a 16-row i-tile; loop all 384 j-tiles,
// 3 WMMA chains per tile (pos, feat, nz2·nz2); fused argmax + denominator.
// ---------------------------------------------------------------------------
__global__ __launch_bounds__(256) void k_loss(
    const unsigned short* __restrict__ posb,
    const unsigned short* __restrict__ featb,
    const unsigned short* __restrict__ nz1b,
    const unsigned short* __restrict__ nz2b,
    const float* __restrict__ alpha,
    float* __restrict__ out)
{
  __shared__ float s_top[8][16];
  __shared__ float s_den[8][16];
  __shared__ int   s_idx[8][16];

  int lane = threadIdx.x & 31;
  int wave = threadIdx.x >> 5;
  int i0 = (blockIdx.x * 8 + wave) * 16;
  float sg = 1.f / (1.f + __expf(-alpha[0]));

  float den_acc[8], best[8];
  int besti[8];
#pragma unroll
  for (int r = 0; r < 8; ++r) { den_acc[r] = 0.f; best[r] = -1e30f; besti[r] = 0; }

  const v8f vz = {0.f, 0.f, 0.f, 0.f, 0.f, 0.f, 0.f, 0.f};
  int n = lane & 15;

  for (int jt = 0; jt < N_NODES / 16; ++jt) {
    int j0 = jt * 16;
    v8f dpp = vz, dff = vz, drr = vz;
#pragma unroll
    for (int ks = 0; ks < 8; ++ks) {
      int k0 = ks * 32;
      dpp = wmma_bf16(frag_a(posb  + (size_t)i0 * 256 + k0, 256, lane),
                      frag_b(posb  + (size_t)j0 * 256 + k0, 256, lane), dpp);
      dff = wmma_bf16(frag_a(featb + (size_t)i0 * 256 + k0, 256, lane),
                      frag_b(featb + (size_t)j0 * 256 + k0, 256, lane), dff);
      drr = wmma_bf16(frag_a(nz2b  + (size_t)i0 * 256 + k0, 256, lane),
                      frag_b(nz2b  + (size_t)j0 * 256 + k0, 256, lane), drr);
    }
#pragma unroll
    for (int r = 0; r < 8; ++r) {
      float pm = sg * dpp[r] + (1.f - sg) * dff[r];
      den_acc[r] += __expf(drr[r] * INV_TEMP) * (1.f - pm);
      if (pm > best[r]) { best[r] = pm; besti[r] = j0 + n; }
    }
  }

  // reduce over the 16 columns (lanes within each 16-lane half)
#pragma unroll
  for (int m = 1; m < 16; m <<= 1) {
#pragma unroll
    for (int r = 0; r < 8; ++r) {
      den_acc[r] += __shfl_xor(den_acc[r], m, 16);
      float ov = __shfl_xor(best[r], m, 16);
      int   oi = __shfl_xor(besti[r], m, 16);
      if (ov > best[r] || (ov == best[r] && oi < besti[r])) {
        best[r] = ov; besti[r] = oi;
      }
    }
  }
  if ((lane & 15) == 0) {
    int hf = lane >> 4;
#pragma unroll
    for (int r = 0; r < 8; ++r) {
      int m = hf * 8 + r;
      s_top[wave][m] = best[r];
      s_den[wave][m] = den_acc[r];
      s_idx[wave][m] = besti[r];
    }
  }
  __syncthreads();

  // between[i, idx[i]] : 2 lanes per row, 128 elements each
  int m = lane >> 1;
  int idx = s_idx[wave][m];
  const unsigned short* x = nz2b + (size_t)(i0 + m) * 256 + (lane & 1) * 128;
  const unsigned short* y = nz1b + (size_t)idx * 256 + (lane & 1) * 128;
  float dot = 0.f;
  for (int d = 0; d < 128; ++d) dot += bf2f(x[d]) * bf2f(y[d]);
  dot += __shfl_xor(dot, 1, 32);
  if ((lane & 1) == 0) {
    float top = s_top[wave][m];
    float numer = top * top * __expf(dot * INV_TEMP);
    float loss = -(__logf(numer) - __logf(s_den[wave][m]));
    atomicAdd(out, loss * (1.f / (float)N_NODES));
  }
}

// ---------------------------------------------------------------------------
// Host orchestration
// ---------------------------------------------------------------------------
extern "C" void kernel_launch(void* const* d_in, const int* in_sizes, int n_in,
                              void* d_out, int out_size, void* d_ws, size_t ws_size,
                              hipStream_t stream) {
  (void)n_in; (void)out_size; (void)ws_size;
  constexpr int N = N_NODES;
  const float* feat1 = (const float*)d_in[0];
  const float* feat2 = (const float*)d_in[1];
  const float* pos_  = (const float*)d_in[2];
  const float* featc = (const float*)d_in[3];
  const float* W1    = (const float*)d_in[4];
  const float* al1   = (const float*)d_in[5];
  const float* ar1   = (const float*)d_in[6];
  const float* b1    = (const float*)d_in[7];
  const float* res1  = (const float*)d_in[8];
  const float* g1    = (const float*)d_in[9];
  const float* be1   = (const float*)d_in[10];
  const float* W2    = (const float*)d_in[11];
  const float* al2   = (const float*)d_in[12];
  const float* ar2   = (const float*)d_in[13];
  const float* b2    = (const float*)d_in[14];
  const float* res2  = (const float*)d_in[15];
  const float* Wp    = (const float*)d_in[16];
  const float* bp    = (const float*)d_in[17];
  const float* alpha = (const float*)d_in[18];
  const int*   srcI  = (const int*)d_in[19];
  const int*   dstI  = (const int*)d_in[20];
  const int Etot = in_sizes[19];

  // bump allocator over d_ws
  char* base = (char*)d_ws;
  size_t off = 0;
  auto alloc = [&](size_t bytes) -> void* {
    void* r = base + off;
    off += (bytes + 255) & ~(size_t)255;
    return r;
  };
  typedef unsigned short bfp;
  bfp* W1b   = (bfp*)alloc((size_t)512 * 1024 * 2);
  bfp* res1b = (bfp*)alloc((size_t)512 * 1024 * 2);
  bfp* W2b   = (bfp*)alloc((size_t)1024 * 1536 * 2);
  bfp* res2b = (bfp*)alloc((size_t)1024 * 1536 * 2);
  bfp* Wpb   = (bfp*)alloc((size_t)256 * 256 * 2);
  bfp* posb  = (bfp*)alloc((size_t)N * 256 * 2);
  bfp* featb = (bfp*)alloc((size_t)N * 256 * 2);
  bfp* nz1b  = (bfp*)alloc((size_t)N * 256 * 2);
  bfp* nz2b  = (bfp*)alloc((size_t)N * 256 * 2);
  bfp* Xb    = (bfp*)alloc((size_t)N * 512 * 2);
  bfp* H1b   = (bfp*)alloc((size_t)N * 1024 * 2);
  bfp* hencb = (bfp*)alloc((size_t)N * 256 * 2);
  float* F1   = (float*)alloc((size_t)N * 1024 * 4);
  float* O1   = (float*)alloc((size_t)N * 1024 * 4);
  float* F2   = (float*)alloc((size_t)N * 1536 * 4);
  float* O2   = (float*)alloc((size_t)N * 1536 * 4);
  float* elv  = (float*)alloc((size_t)N * 6 * 4);
  float* erv  = (float*)alloc((size_t)N * 6 * 4);
  float* emax = (float*)alloc((size_t)N * 6 * 4);
  float* den  = (float*)alloc((size_t)N * 6 * 4);
  float* Z1   = (float*)alloc((size_t)N * 256 * 4);
  float* Z2   = (float*)alloc((size_t)N * 256 * 4);
  float* bnsum = (float*)alloc(1024 * 4);
  float* bnss  = (float*)alloc(1024 * 4);

  auto cvt = [&](const float* in, bfp* o, int n) {
    k_cvt_bf16<<<(n + 255) / 256, 256, 0, stream>>>(in, o, n);
  };
  auto fill = [&](float* p, float v, int n) {
    k_fill<<<(n + 255) / 256, 256, 0, stream>>>(p, v, n);
  };

  // one-time (per call) weight / constant conversions
  cvt(W1, W1b, 512 * 1024);
  cvt(res1, res1b, 512 * 1024);
  cvt(W2, W2b, 1024 * 1536);
  cvt(res2, res2b, 1024 * 1536);
  cvt(Wp, Wpb, 256 * 256);
  cvt(pos_, posb, N * 256);
  cvt(featc, featb, N * 256);

  auto gat_edges = [&](const float* F, float* O, const float* al, const float* ar,
                       int H) {
    k_elr<<<N, 32 * H, 0, stream>>>(F, al, ar, elv, erv, H);
    fill(emax, -1e30f, N * H);
    fill(den, 0.f, N * H);
    int tH = Etot * H;
    k_edge_max<<<(tH + 255) / 256, 256, 0, stream>>>(elv, erv, srcI, dstI, emax, Etot, H);
    k_edge_den<<<(tH + 255) / 256, 256, 0, stream>>>(elv, erv, srcI, dstI, emax, den, Etot, H);
    k_edge_agg<<<Etot, 256, 0, stream>>>(elv, erv, emax, den, srcI, dstI, F, O, H);
  };

  auto encode = [&](const float* x, float* Z) {
    cvt(x, Xb, N * 512);
    // layer 1
    dim3 grd1(1024 / 128, N / 128);
    k_gemm<<<grd1, 256, 0, stream>>>(Xb, W1b, F1, (const float*)nullptr, 512, 1024);
    k_gemm<<<grd1, 256, 0, stream>>>(Xb, res1b, O1, b1, 512, 1024);
    gat_edges(F1, O1, al1, ar1, NH1);
    // batchnorm + ELU
    fill(bnsum, 0.f, 1024);
    fill(bnss, 0.f, 1024);
    k_bn_stats<<<N / 4, 256, 0, stream>>>(O1, bnsum, bnss);
    k_bn_apply<<<N, 256, 0, stream>>>(O1, bnsum, bnss, g1, be1, H1b);
    // layer 2
    dim3 grd2(1536 / 128, N / 128);
    k_gemm<<<grd2, 256, 0, stream>>>(H1b, W2b, F2, (const float*)nullptr, 1024, 1536);
    k_gemm<<<grd2, 256, 0, stream>>>(H1b, res2b, O2, b2, 1024, 1536);
    gat_edges(F2, O2, al2, ar2, NH2);
    // ELU + head-mean, projection
    k_elu_mean<<<N, 256, 0, stream>>>(O2, hencb);
    dim3 grdp(256 / 128, N / 128);
    k_gemm<<<grdp, 256, 0, stream>>>(hencb, Wpb, Z, bp, 256, 256);
  };

  encode(feat1, Z1);
  encode(feat2, Z2);

  k_l2norm<<<N, 256, 0, stream>>>(Z1, nz1b);
  k_l2norm<<<N, 256, 0, stream>>>(Z2, nz2b);

  fill((float*)d_out, 0.f, 1);
  k_loss<<<N / 128, 256, 0, stream>>>(posb, featb, nz1b, nz2b, alpha, (float*)d_out);
}
